// LRU_7619271983134
// MI455X (gfx1250) — matrix operand order repaired and checked by reading, not compiled, not run on hardware
//
#include <hip/hip_runtime.h>
#include <math.h>

typedef __attribute__((ext_vector_type(16))) _Float16 v16h;
typedef __attribute__((ext_vector_type(8)))  float    v8f;
typedef __attribute__((ext_vector_type(4)))  float    f4;

#define LSEQ   8192
#define NDIM   512
#define HDIM   512
#define CHUNK  128
#define NCHUNK (LSEQ / CHUNK)

// ---------------------------------------------------------------------------
// Phase 0: Lambda = exp(-exp(nu_log) + i*exp(theta_log)), scale = exp(theta_log)
// ---------------------------------------------------------------------------
__global__ __launch_bounds__(NDIM) void lru_prep(
    const float* __restrict__ nu_log, const float* __restrict__ theta_log,
    float* __restrict__ lam_re, float* __restrict__ lam_im,
    float* __restrict__ exp_theta)
{
    int n = threadIdx.x;
    float theta = expf(theta_log[n]);
    float nu    = expf(nu_log[n]);
    float mod   = expf(-nu);
    exp_theta[n] = theta;
    lam_re[n] = mod * cosf(theta);
    lam_im[n] = mod * sinf(theta);
}

// ---------------------------------------------------------------------------
// Phase 1: Bu_re/Bu_im (L,N) = u (L,H) @ (B_{re,im} * exp_theta[n])^T  via WMMA
// One wave -> one 16x16 tile; K loop over H in steps of 32.
// ---------------------------------------------------------------------------
__global__ __launch_bounds__(256) void lru_gemm_bu(
    const float* __restrict__ u,        // (L,H)
    const float* __restrict__ B_re,     // (N,H)
    const float* __restrict__ B_im,     // (N,H)
    const float* __restrict__ exp_theta,// (N)
    float* __restrict__ Bu_re,          // (L,N)
    float* __restrict__ Bu_im)          // (L,N)
{
    const int lane = threadIdx.x & 31;
    const int wave = threadIdx.x >> 5;
    const int tile = blockIdx.x * 8 + wave;      // 16384 tiles total
    const int tm   = tile >> 5;                  // 512 row tiles (L/16)
    const int tn   = tile & 31;                  // 32 col tiles (N/16)
    const int l16  = lane & 15;
    const int half = lane >> 4;

    const int arow = tm * 16 + l16;
    const int bcol = tn * 16 + l16;
    const float* __restrict__ urow  = u    + (size_t)arow * HDIM;
    const float* __restrict__ brrow = B_re + (size_t)bcol * HDIM;
    const float* __restrict__ birow = B_im + (size_t)bcol * HDIM;
    const float  scale = exp_theta[bcol];

    v8f acc_re = {};
    v8f acc_im = {};
    for (int k0 = 0; k0 < HDIM; k0 += 32) {
        // A operand: 16 contiguous K values of row `arow` (matches B order)
        const float* a1 = urow + k0 + half * 16;
        const float* a2 = a1 + 8;
        f4 av0 = *(const f4*)(a1);
        f4 av1 = *(const f4*)(a1 + 4);
        f4 av2 = *(const f4*)(a2);
        f4 av3 = *(const f4*)(a2 + 4);
        v16h a;
#pragma unroll
        for (int i = 0; i < 4; ++i) {
            a[i]      = (_Float16)av0[i];
            a[4 + i]  = (_Float16)av1[i];
            a[8 + i]  = (_Float16)av2[i];
            a[12 + i] = (_Float16)av3[i];
        }
        // B operand: 16 contiguous K values of row `bcol` (theta scale folded in)
        const float* b1 = brrow + k0 + half * 16;
        const float* b2 = birow + k0 + half * 16;
        v16h br, bi;
#pragma unroll
        for (int j = 0; j < 4; ++j) {
            f4 vr = *(const f4*)(b1 + 4 * j);
            f4 vi = *(const f4*)(b2 + 4 * j);
#pragma unroll
            for (int i = 0; i < 4; ++i) {
                br[4 * j + i] = (_Float16)(vr[i] * scale);
                bi[4 * j + i] = (_Float16)(vi[i] * scale);
            }
        }
        if (k0 + 32 < HDIM) {
            __builtin_prefetch(a1 + 32, 0, 3);
            __builtin_prefetch(b1 + 32, 0, 3);
            __builtin_prefetch(b2 + 32, 0, 3);
        }
        acc_re = __builtin_amdgcn_wmma_f32_16x16x32_f16(
            false, a, false, br, (short)0, acc_re, false, false);
        acc_im = __builtin_amdgcn_wmma_f32_16x16x32_f16(
            false, a, false, bi, (short)0, acc_im, false, false);
    }
    // C/D layout: lane writes column l16, rows v + 8*half
#pragma unroll
    for (int v = 0; v < 8; ++v) {
        int row = tm * 16 + v + half * 8;
        Bu_re[(size_t)row * NDIM + bcol] = acc_re[v];
        Bu_im[(size_t)row * NDIM + bcol] = acc_im[v];
    }
}

// ---------------------------------------------------------------------------
// Phase 2a: per-chunk local scans, in place over Bu. One thread per channel.
// ---------------------------------------------------------------------------
__global__ __launch_bounds__(NDIM) void lru_scan_local(
    float* __restrict__ Bu_re, float* __restrict__ Bu_im,
    const float* __restrict__ lam_re, const float* __restrict__ lam_im,
    float* __restrict__ last_re, float* __restrict__ last_im)
{
    const int n = threadIdx.x;
    const int c = blockIdx.x;
    const float lr = lam_re[n], li = lam_im[n];
    float ar = 0.f, ai = 0.f;
    const int t0 = c * CHUNK;
    for (int t = t0; t < t0 + CHUNK; ++t) {
        size_t idx = (size_t)t * NDIM + n;
        float br = Bu_re[idx];
        float bi = Bu_im[idx];
        float nr = fmaf(lr, ar, fmaf(-li, ai, br));
        float ni = fmaf(lr, ai, fmaf( li, ar, bi));
        ar = nr; ai = ni;
        Bu_re[idx] = ar;
        Bu_im[idx] = ai;
    }
    last_re[c * NDIM + n] = ar;
    last_im[c * NDIM + n] = ai;
}

// ---------------------------------------------------------------------------
// Phase 2b: scan of chunk carries. carry[c] = Lam^CHUNK * carry[c-1] + last[c-1]
// Lam^128 via 7 complex squarings. Single block, one thread per channel.
// ---------------------------------------------------------------------------
__global__ __launch_bounds__(NDIM) void lru_scan_carry(
    const float* __restrict__ last_re, const float* __restrict__ last_im,
    const float* __restrict__ lam_re, const float* __restrict__ lam_im,
    float* __restrict__ carry_re, float* __restrict__ carry_im)
{
    const int n = threadIdx.x;
    float pr = lam_re[n], pi = lam_im[n];
#pragma unroll
    for (int i = 0; i < 7; ++i) {          // Lambda^(2^7) = Lambda^CHUNK
        float nr = pr * pr - pi * pi;
        float ni = 2.f * pr * pi;
        pr = nr; pi = ni;
    }
    float cr = 0.f, ci = 0.f;
    carry_re[n] = 0.f;
    carry_im[n] = 0.f;
    for (int c = 1; c < NCHUNK; ++c) {
        float fr = last_re[(c - 1) * NDIM + n];
        float fi = last_im[(c - 1) * NDIM + n];
        float nr = fmaf(pr, cr, fmaf(-pi, ci, fr));
        float ni = fmaf(pr, ci, fmaf( pi, cr, fi));
        cr = nr; ci = ni;
        carry_re[c * NDIM + n] = cr;
        carry_im[c * NDIM + n] = ci;
    }
}

// ---------------------------------------------------------------------------
// Phase 2c: fixup. states[t in chunk c] += Lambda^(t-t0+1) * carry[c]
// Chunk 0 needs no fixup -> grid covers chunks 1..NCHUNK-1.
// ---------------------------------------------------------------------------
__global__ __launch_bounds__(NDIM) void lru_scan_fixup(
    float* __restrict__ Bu_re, float* __restrict__ Bu_im,
    const float* __restrict__ lam_re, const float* __restrict__ lam_im,
    const float* __restrict__ carry_re, const float* __restrict__ carry_im)
{
    const int n = threadIdx.x;
    const int c = blockIdx.x + 1;
    const float cr = carry_re[c * NDIM + n];
    const float ci = carry_im[c * NDIM + n];
    const float lr = lam_re[n], li = lam_im[n];
    float pr = lr, pi = li;
    const int t0 = c * CHUNK;
    for (int t = t0; t < t0 + CHUNK; ++t) {
        size_t idx = (size_t)t * NDIM + n;
        Bu_re[idx] = fmaf(pr, cr, fmaf(-pi, ci, Bu_re[idx]));
        Bu_im[idx] = fmaf(pr, ci, fmaf( pi, cr, Bu_im[idx]));
        float nr = pr * lr - pi * li;
        float ni = pr * li + pi * lr;
        pr = nr; pi = ni;
    }
}

// ---------------------------------------------------------------------------
// Phase 3: y = st_re @ C_re^T - st_im @ C_im^T + D*u   via WMMA
// (-st_im) folded into the f16 conversion of the A operand (f16 WMMA has no
//  A/B NEG modifier per ISA NEG table).
// ---------------------------------------------------------------------------
__global__ __launch_bounds__(256) void lru_gemm_out(
    const float* __restrict__ st_re,  // (L,N)
    const float* __restrict__ st_im,  // (L,N)
    const float* __restrict__ C_re,   // (H,N)
    const float* __restrict__ C_im,   // (H,N)
    const float* __restrict__ Dvec,   // (H)
    const float* __restrict__ u,      // (L,H)
    float* __restrict__ y)            // (L,H)
{
    const int lane = threadIdx.x & 31;
    const int wave = threadIdx.x >> 5;
    const int tile = blockIdx.x * 8 + wave;
    const int tm   = tile >> 5;                  // 512 row tiles (L/16)
    const int tn   = tile & 31;                  // 32 col tiles (H/16)
    const int l16  = lane & 15;
    const int half = lane >> 4;

    const int arow = tm * 16 + l16;
    const int bcol = tn * 16 + l16;
    const float* __restrict__ srrow = st_re + (size_t)arow * NDIM;
    const float* __restrict__ sirow = st_im + (size_t)arow * NDIM;
    const float* __restrict__ crrow = C_re  + (size_t)bcol * NDIM;
    const float* __restrict__ cirow = C_im  + (size_t)bcol * NDIM;

    v8f acc = {};
    for (int k0 = 0; k0 < NDIM; k0 += 32) {
        const float* ar1 = srrow + k0 + half * 16;
        const float* ai1 = sirow + k0 + half * 16;
        v16h a_re, a_imn;
#pragma unroll
        for (int j = 0; j < 2; ++j) {
            f4 r0 = *(const f4*)(ar1 + 8 * j);
            f4 r1 = *(const f4*)(ar1 + 8 * j + 4);
            f4 i0 = *(const f4*)(ai1 + 8 * j);
            f4 i1 = *(const f4*)(ai1 + 8 * j + 4);
#pragma unroll
            for (int i = 0; i < 4; ++i) {
                a_re [8 * j + i]     = (_Float16)r0[i];
                a_re [8 * j + 4 + i] = (_Float16)r1[i];
                a_imn[8 * j + i]     = (_Float16)(-i0[i]);
                a_imn[8 * j + 4 + i] = (_Float16)(-i1[i]);
            }
        }
        const float* b1 = crrow + k0 + half * 16;
        const float* b2 = cirow + k0 + half * 16;
        v16h bcr, bci;
#pragma unroll
        for (int j = 0; j < 4; ++j) {
            f4 vr = *(const f4*)(b1 + 4 * j);
            f4 vi = *(const f4*)(b2 + 4 * j);
#pragma unroll
            for (int i = 0; i < 4; ++i) {
                bcr[4 * j + i] = (_Float16)vr[i];
                bci[4 * j + i] = (_Float16)vi[i];
            }
        }
        if (k0 + 32 < NDIM) {
            __builtin_prefetch(ar1 + 32, 0, 3);
            __builtin_prefetch(ai1 + 32, 0, 3);
            __builtin_prefetch(b1 + 32, 0, 3);
            __builtin_prefetch(b2 + 32, 0, 3);
        }
        acc = __builtin_amdgcn_wmma_f32_16x16x32_f16(
            false, a_re, false, bcr, (short)0, acc, false, false);
        acc = __builtin_amdgcn_wmma_f32_16x16x32_f16(
            false, a_imn, false, bci, (short)0, acc, false, false);
    }
    const float d = Dvec[bcol];
#pragma unroll
    for (int v = 0; v < 8; ++v) {
        int row = tm * 16 + v + half * 8;
        size_t idx = (size_t)row * HDIM + bcol;
        y[idx] = acc[v] + d * u[idx];
    }
}

// ---------------------------------------------------------------------------
extern "C" void kernel_launch(void* const* d_in, const int* in_sizes, int n_in,
                              void* d_out, int out_size, void* d_ws, size_t ws_size,
                              hipStream_t stream) {
    (void)in_sizes; (void)n_in; (void)out_size; (void)ws_size;
    const float* u         = (const float*)d_in[0];  // (L,H)
    const float* nu_log    = (const float*)d_in[1];  // (N)
    const float* theta_log = (const float*)d_in[2];  // (N)
    // d_in[3] = gamma_log: unused by the reference math path (it scales B by
    // exp(theta_log), as written in the source).
    const float* B_re      = (const float*)d_in[4];  // (N,H)
    const float* B_im      = (const float*)d_in[5];  // (N,H)
    const float* C_re      = (const float*)d_in[6];  // (H,N)
    const float* C_im      = (const float*)d_in[7];  // (H,N)
    const float* Dvec      = (const float*)d_in[8];  // (H)
    float* y = (float*)d_out;

    // Workspace layout (floats): ~34 MB total
    float* f        = (float*)d_ws;
    float* Bu_re    = f;                              // L*N
    float* Bu_im    = Bu_re    + (size_t)LSEQ * NDIM; // L*N
    float* last_re  = Bu_im    + (size_t)LSEQ * NDIM; // NCHUNK*N
    float* last_im  = last_re  + NCHUNK * NDIM;
    float* carry_re = last_im  + NCHUNK * NDIM;
    float* carry_im = carry_re + NCHUNK * NDIM;
    float* lam_re   = carry_im + NCHUNK * NDIM;       // N
    float* lam_im   = lam_re   + NDIM;
    float* exp_th   = lam_im   + NDIM;

    const int tiles  = (LSEQ / 16) * (NDIM / 16);     // 16384
    const int blocks = tiles / 8;                     // 8 waves (tiles) / block

    lru_prep<<<1, NDIM, 0, stream>>>(nu_log, theta_log, lam_re, lam_im, exp_th);

    lru_gemm_bu<<<blocks, 256, 0, stream>>>(u, B_re, B_im, exp_th, Bu_re, Bu_im);

    lru_scan_local<<<NCHUNK, NDIM, 0, stream>>>(Bu_re, Bu_im, lam_re, lam_im,
                                                last_re, last_im);
    lru_scan_carry<<<1, NDIM, 0, stream>>>(last_re, last_im, lam_re, lam_im,
                                           carry_re, carry_im);
    lru_scan_fixup<<<NCHUNK - 1, NDIM, 0, stream>>>(Bu_re, Bu_im, lam_re, lam_im,
                                                    carry_re, carry_im);

    lru_gemm_out<<<blocks, 256, 0, stream>>>(Bu_re, Bu_im, C_re, C_im, Dvec, u, y);
}
